// GCNLayer_77713138253972
// MI455X (gfx1250) — matrix operand order repaired and checked by reading, not compiled, not run on hardware
//
#include <hip/hip_runtime.h>

#define BATCH 16
#define NN    1024
#define FF    128
#define ROWS  128   // rows per block in kernels 2 & 3
#define KC    128   // K-chunk for kernel 3
#define LDSP  136   // padded LDS row stride (halves), keeps 16B alignment

typedef __attribute__((ext_vector_type(16))) __bf16 v16bf;
typedef __attribute__((ext_vector_type(8)))  float  v8f;

struct __align__(16) BF8 { __bf16 h[8]; };

// A-matrix fragment (16-bit 16x32): lanes 0-15: K = {k0..k0+7, k0+16..k0+23}
//                                   lanes16-31: K = {k0+8..k0+15, k0+24..k0+31}
__device__ __forceinline__ v16bf frag_A(const __bf16* row, int k0, int hi) {
    const int o = k0 + (hi ? 8 : 0);
    BF8 g0 = *reinterpret_cast<const BF8*>(row + o);
    BF8 g1 = *reinterpret_cast<const BF8*>(row + o + 16);
    v16bf r;
#pragma unroll
    for (int i = 0; i < 8; ++i) { r[i] = g0.h[i]; r[i + 8] = g1.h[i]; }
    return r;
}

// B-matrix fragment (16-bit 32x16): lanes 0-15 hold K = {k0..k0+7, k0+16..k0+23},
//                                   lanes16-31 hold K = {k0+8..k0+15, k0+24..k0+31} (per column)
__device__ __forceinline__ v16bf frag_B(const __bf16* col, int k0, int hi) {
    const int o = k0 + (hi ? 8 : 0);
    BF8 g0 = *reinterpret_cast<const BF8*>(col + o);
    BF8 g1 = *reinterpret_cast<const BF8*>(col + o + 16);
    v16bf r;
#pragma unroll
    for (int i = 0; i < 8; ++i) { r[i] = g0.h[i]; r[i + 8] = g1.h[i]; }
    return r;
}

// CDNA5 async global->LDS copy (ASYNCcnt-tracked DMA, no VGPR round-trip).
__device__ __forceinline__ void async_copy_b128(unsigned lds_off, unsigned long long gaddr) {
    asm volatile("global_load_async_to_lds_b128 %0, %1, off"
                 :: "v"(lds_off), "v"(gaddr) : "memory");
}
__device__ __forceinline__ void wait_async0() {
    asm volatile("s_wait_asynccnt 0" ::: "memory");
}

// ---------------- Kernel 1: ds[b,n] = rsqrt(1 + rowsum(A[b,n,:])) ----------------
__global__ __launch_bounds__(256)
void gcn_rowsum(const float* __restrict__ A, float* __restrict__ ds) {
    const int tid  = threadIdx.x;
    const int wave = tid >> 5;
    const int lane = tid & 31;
    const int row  = blockIdx.x * 8 + wave;          // flat row in [0, B*N)
    const float4* a4 = reinterpret_cast<const float4*>(A + (size_t)row * NN);
    float s = 0.f;
#pragma unroll
    for (int i = 0; i < 8; ++i) {
        float4 v = a4[i * 32 + lane];
        s += v.x + v.y + v.z + v.w;
    }
#pragma unroll
    for (int off = 16; off > 0; off >>= 1) s += __shfl_xor(s, off, 32);
    if (lane == 0) ds[row] = rsqrtf(1.0f + s);
}

// ---- Kernel 2: Yt[b,g,m] = bf16( ds[b,m] * sum_f X[b,m,f]*W[f,g] )  (TRANSPOSED) --
__global__ __launch_bounds__(256)
void gcn_xw(const float* __restrict__ X, const float* __restrict__ W,
            const float* __restrict__ ds, __bf16* __restrict__ Yt) {
    __shared__ __align__(16) __bf16 Xs[ROWS][LDSP];   // ds-scaled X rows, K = f
    __shared__ __align__(16) __bf16 Wt[FF][LDSP];     // W transposed: [g][f]

    const int b    = blockIdx.x / (NN / ROWS);
    const int m0   = (blockIdx.x % (NN / ROWS)) * ROWS;
    const int tid  = threadIdx.x;
    const int wave = tid >> 5;
    const int lane = tid & 31;
    const float* Xb = X + (size_t)b * NN * FF;

    // stage ds-scaled X slab (fp32 -> bf16)
    for (int i = tid; i < ROWS * (FF / 4); i += 256) {
        const int r = i >> 5, c4 = i & 31;
        float4 v = *reinterpret_cast<const float4*>(Xb + (size_t)(m0 + r) * FF + (c4 << 2));
        const float s = ds[b * NN + m0 + r];
        Xs[r][(c4 << 2) + 0] = (__bf16)(v.x * s);
        Xs[r][(c4 << 2) + 1] = (__bf16)(v.y * s);
        Xs[r][(c4 << 2) + 2] = (__bf16)(v.z * s);
        Xs[r][(c4 << 2) + 3] = (__bf16)(v.w * s);
    }
    // stage W transposed
    for (int i = tid; i < FF * FF; i += 256) {
        const int f = i >> 7, g = i & 127;
        Wt[g][f] = (__bf16)W[i];
    }
    __syncthreads();

    const int col   = lane & 15;
    const int hi    = lane >> 4;
    const int rtile = wave << 4;                      // 16 rows per wave

    v8f acc[8];
#pragma unroll
    for (int t = 0; t < 8; ++t) acc[t] = (v8f){0.f,0.f,0.f,0.f,0.f,0.f,0.f,0.f};

#pragma unroll
    for (int kk = 0; kk < FF; kk += 32) {
        const v16bf af = frag_A(&Xs[rtile + col][0], kk, hi);
#pragma unroll
        for (int t = 0; t < 8; ++t) {
            const v16bf bfrag = frag_B(&Wt[(t << 4) + col][0], kk, hi);
            acc[t] = __builtin_amdgcn_wmma_f32_16x16x32_bf16(
                false, af, false, bfrag, (short)0, acc[t], false, false);
        }
    }

    // accumulator rows v=0..7 are consecutive m -> one 16B store per tile
    const int rbase = m0 + rtile + (hi << 3);
#pragma unroll
    for (int t = 0; t < 8; ++t) {
        const int g = (t << 4) + col;
        BF8 pack;
#pragma unroll
        for (int v = 0; v < 8; ++v) pack.h[v] = (__bf16)acc[t][v];
        *reinterpret_cast<BF8*>(Yt + ((size_t)b * FF + g) * NN + rbase) = pack;
    }
}

// -- Kernel 3: out[b,n,g] = ds[b,n]*(sum_m A[b,n,m]*Y[b,m,g] + Y[b,n,g]), Y from Yt --
__global__ __launch_bounds__(256)
void gcn_aggr(const float* __restrict__ A, const __bf16* __restrict__ Yt,
              const float* __restrict__ ds, float* __restrict__ out) {
    __shared__ __align__(16) __bf16 As[ROWS][LDSP];   // bf16 A chunk, K = m
    __shared__ __align__(16) __bf16 Ys[FF][LDSP];     // Yt chunk: [g][m] (already transposed)

    const int b     = blockIdx.x / (NN / ROWS);
    const int r0blk = (blockIdx.x % (NN / ROWS)) * ROWS;
    const int tid   = threadIdx.x;
    const int wave  = tid >> 5;
    const int lane  = tid & 31;
    const float*  Ab  = A  + (size_t)b * NN * NN;
    const __bf16* Ytb = Yt + (size_t)b * FF * NN;

    const int col = lane & 15;
    const int hi  = lane >> 4;

    const unsigned ys_base = (unsigned)(unsigned long long)(uintptr_t)&Ys[0][0];

    v8f acc[8];
#pragma unroll
    for (int t = 0; t < 8; ++t) acc[t] = (v8f){0.f,0.f,0.f,0.f,0.f,0.f,0.f,0.f};

    for (int kc = 0; kc < NN; kc += KC) {
        if (kc) __syncthreads();
        // async DMA the Yt chunk (pure copy, 16B per lane-op): Ys[g][0..KC-1]
        for (int i = tid; i < FF * (KC / 8); i += 256) {          // 2048 segments
            const int g = i >> 4, seg = i & 15;
            const unsigned lds_off = ys_base + (unsigned)(g * (LDSP * 2) + seg * 16);
            const unsigned long long ga =
                (unsigned long long)(uintptr_t)(Ytb + (size_t)g * NN + kc + seg * 8);
            async_copy_b128(lds_off, ga);
        }
        // stage A chunk (fp32 -> bf16) through VGPRs (overlaps the async DMA)
        for (int i = tid; i < ROWS * (KC / 4); i += 256) {
            const int r = i >> 5, c4 = i & 31;
            float4 v = *reinterpret_cast<const float4*>(
                Ab + (size_t)(r0blk + r) * NN + kc + (c4 << 2));
            As[r][(c4 << 2) + 0] = (__bf16)v.x;
            As[r][(c4 << 2) + 1] = (__bf16)v.y;
            As[r][(c4 << 2) + 2] = (__bf16)v.z;
            As[r][(c4 << 2) + 3] = (__bf16)v.w;
        }
        wait_async0();
        __syncthreads();

        const int rtile = wave << 4;
#pragma unroll
        for (int kk = 0; kk < KC; kk += 32) {
            const v16bf af = frag_A(&As[rtile + col][0], kk, hi);
#pragma unroll
            for (int t = 0; t < 8; ++t) {
                const v16bf bfrag = frag_B(&Ys[(t << 4) + col][0], kk, hi);
                acc[t] = __builtin_amdgcn_wmma_f32_16x16x32_bf16(
                    false, af, false, bfrag, (short)0, acc[t], false, false);
            }
        }
    }

    // epilogue: identity term (one 16B load per tile from Yt), scale by ds[n]
    const int rbase = r0blk + (wave << 4) + (hi << 3);
#pragma unroll
    for (int t = 0; t < 8; ++t) {
        const int g = (t << 4) + col;
        const BF8 yid = *reinterpret_cast<const BF8*>(Ytb + (size_t)g * NN + rbase);
#pragma unroll
        for (int v = 0; v < 8; ++v) {
            const int n = rbase + v;
            const float val = acc[t][v] + (float)yid.h[v];
            out[((size_t)b * NN + n) * FF + g] = ds[b * NN + n] * val;
        }
    }
}

extern "C" void kernel_launch(void* const* d_in, const int* in_sizes, int n_in,
                              void* d_out, int out_size, void* d_ws, size_t ws_size,
                              hipStream_t stream) {
    const float* X = (const float*)d_in[0];
    const float* A = (const float*)d_in[1];
    const float* W = (const float*)d_in[2];
    float* out = (float*)d_out;

    float*  ds = (float*)d_ws;                                  // B*N fp32  (64 KB)
    __bf16* Yt = (__bf16*)((char*)d_ws + 64 * 1024);            // B*F*N bf16 (4 MB), transposed

    gcn_rowsum<<<dim3(BATCH * NN / 8), dim3(256), 0, stream>>>(A, ds);
    gcn_xw    <<<dim3(BATCH * (NN / ROWS)), dim3(256), 0, stream>>>(X, W, ds, Yt);
    gcn_aggr  <<<dim3(BATCH * (NN / ROWS)), dim3(256), 0, stream>>>(A, Yt, ds, out);
}
